// CoOccurrenceEncoder_28802050686991
// MI455X (gfx1250) — matrix operand order repaired and checked by reading, not compiled, not run on hardware
//
#include <hip/hip_runtime.h>
#include <hip/hip_bf16.h>

// CDNA5 / gfx1250: wave32, WMMA 16x16x32 bf16 -> f32
typedef __attribute__((ext_vector_type(16))) __bf16 v16bf;
typedef __attribute__((ext_vector_type(8)))  float  v8f;

#define D      128
#define NNODES 512
#define BATCH  4

union BF16x16 {
    v16bf v;
    uint4 u[2];
};

// ---------------------------------------------------------------------------
// Kernel 1: hi' = nf @ W1[:D] + b1 ;  hj = nf @ W1[D:]   (all f32, tiny GEMM)
// grid = B*N blocks of D threads; thread k owns output column k.
// ---------------------------------------------------------------------------
__global__ __launch_bounds__(D)
void precompute_hihj(const float* __restrict__ nf,
                     const float* __restrict__ W1,   // [2D, D]
                     const float* __restrict__ b1,   // [D]
                     float* __restrict__ hi_out,     // [B*N, D]
                     float* __restrict__ hj_out) {   // [B*N, D]
    __shared__ float x[D];
    const int row = blockIdx.x;          // b*N + n
    const int k   = threadIdx.x;         // 0..127
    x[k] = nf[(size_t)row * D + k];
    __syncthreads();
    float si = 0.f, sj = 0.f;
#pragma unroll 8
    for (int d = 0; d < D; ++d) {
        const float xv = x[d];
        si = fmaf(xv, W1[d * D + k],       si);
        sj = fmaf(xv, W1[(D + d) * D + k], sj);
    }
    hi_out[(size_t)row * D + k] = si + b1[k];
    hj_out[(size_t)row * D + k] = sj;
}

// ---------------------------------------------------------------------------
// Kernel 2: fully fused pair MLP.
// One workgroup per (b,i): out[b,i,j] = sigmoid(relu(relu(hi'_i+hj_j) @ W2 + b2) @ W3 + b3)
// 8 waves x 64 j's each; per 16-row tile: 16x v_wmma_f32_16x16x32_bf16.
// A-fragment (16-bit A 16x32): lane L (g=L>>4, m=L&15) holds, for element e:
//   k = (e < 8) ? g*8 + e : 16 + g*8 + (e-8)
// B-fragment (16-bit B 32x16): lane L (n=L&15, g=L>>4) element e: k = (e < 8) ? g*8 + e : 16 + g*8 + (e-8)
// C layout (f32 16x16): VGPR r: lanes 0-15 -> M=r, lanes 16-31 -> M=r+8, N=lane&15
// ---------------------------------------------------------------------------
__global__ __launch_bounds__(256)
void cooccur_fused(const float* __restrict__ hi_ws,  // [B*N, D]
                   const float* __restrict__ hj_ws,  // [B*N, D]
                   const float* __restrict__ W2,     // [D, 64]
                   const float* __restrict__ b2,     // [64]
                   const float* __restrict__ W3,     // [64]
                   const float* __restrict__ b3,     // [1]
                   float* __restrict__ out) {        // [B*N, N]
    // W2 pre-swizzled to B-fragment layout: [frag=kk*4+nn][chunk 0..1][lane][8 halves]
    __shared__ __align__(16) __bf16 ldsB[16 * 2 * 32 * 8];  // 16 KB
    __shared__ __align__(16) float  lds_hi[D];              // hi' row (with b1)

    const int tid  = threadIdx.x;
    const int lane = tid & 31;
    const int wave = tid >> 5;
    const int g    = lane >> 4;   // half-wave group
    const int lm   = lane & 15;

    // ---- stage hi'_i ----
    if (tid < D) lds_hi[tid] = hi_ws[(size_t)blockIdx.x * D + tid];

    // ---- stage W2 as bf16 WMMA-B fragments (512 lane-slots, 2 per thread) ----
    for (int s = tid; s < 512; s += 256) {
        const int f  = s >> 5;            // fragment id = kk*4 + nn
        const int sl = s & 31;            // lane within fragment
        const int kk = f >> 2, nn = f & 3;
        const int sg = sl >> 4;
        const int n  = nn * 16 + (sl & 15);
#pragma unroll
        for (int e = 0; e < 16; ++e) {
            const int k = kk * 32 + ((e < 8) ? sg * 8 + e : 16 + sg * 8 + (e - 8));
            ldsB[((f * 2 + (e >> 3)) * 32 + sl) * 8 + (e & 7)] =
                (__bf16)W2[k * 64 + n];
        }
    }
    __syncthreads();

    // per-lane epilogue constants: column = nn*16 + lm
    float b2v[4], w3v[4];
#pragma unroll
    for (int nn = 0; nn < 4; ++nn) {
        b2v[nn] = b2[nn * 16 + lm];
        w3v[nn] = W3[nn * 16 + lm];
    }
    const float b3s = b3[0];

    const int    b       = blockIdx.x >> 9;             // batch
    const float* hj_base = hj_ws + (size_t)b * NNODES * D;
    const uint4* bp      = reinterpret_cast<const uint4*>(ldsB);
    float*       orow    = out + (size_t)blockIdx.x * NNODES;

#pragma unroll 1
    for (int t = 0; t < 4; ++t) {
        const int    j0    = wave * 64 + t * 16;
        const float* hjrow = hj_base + (size_t)(j0 + lm) * D;

        // ---- build A fragments: relu(hi' + hj) -> bf16, WMMA A layout ----
        BF16x16 A[4];
#pragma unroll
        for (int kk = 0; kk < 4; ++kk) {
            const int k0 = kk * 32 + g * 8;       // elements 0..7
            const int k1 = kk * 32 + 16 + g * 8;  // elements 8..15
            float hjv[16], hiv[16];
            *reinterpret_cast<float4*>(&hjv[0])  = *reinterpret_cast<const float4*>(hjrow + k0);
            *reinterpret_cast<float4*>(&hjv[4])  = *reinterpret_cast<const float4*>(hjrow + k0 + 4);
            *reinterpret_cast<float4*>(&hjv[8])  = *reinterpret_cast<const float4*>(hjrow + k1);
            *reinterpret_cast<float4*>(&hjv[12]) = *reinterpret_cast<const float4*>(hjrow + k1 + 4);
            *reinterpret_cast<float4*>(&hiv[0])  = *reinterpret_cast<const float4*>(&lds_hi[k0]);
            *reinterpret_cast<float4*>(&hiv[4])  = *reinterpret_cast<const float4*>(&lds_hi[k0 + 4]);
            *reinterpret_cast<float4*>(&hiv[8])  = *reinterpret_cast<const float4*>(&lds_hi[k1]);
            *reinterpret_cast<float4*>(&hiv[12]) = *reinterpret_cast<const float4*>(&lds_hi[k1 + 4]);
#pragma unroll
            for (int e = 0; e < 16; ++e)
                A[kk].v[e] = (__bf16)fmaxf(hiv[e] + hjv[e], 0.f);
        }

        // ---- GEMM vs W2, fused +b2/relu/xW3 epilogue ----
        float lp[8];
#pragma unroll
        for (int r = 0; r < 8; ++r) lp[r] = 0.f;

#pragma unroll
        for (int nn = 0; nn < 4; ++nn) {
            v8f c = {};
#pragma unroll
            for (int kk = 0; kk < 4; ++kk) {
                BF16x16 B;
                const int f = kk * 4 + nn;
                B.u[0] = bp[(f * 2 + 0) * 32 + lane];
                B.u[1] = bp[(f * 2 + 1) * 32 + lane];
                c = __builtin_amdgcn_wmma_f32_16x16x32_bf16(
                        /*neg_a=*/false, A[kk].v,
                        /*neg_b=*/false, B.v,
                        /*c_mod=*/(short)0, c,
                        /*reuse_a=*/false, /*reuse_b=*/false);
            }
            const float bb = b2v[nn];
            const float ww = w3v[nn];
#pragma unroll
            for (int r = 0; r < 8; ++r)
                lp[r] += fmaxf(c[r] + bb, 0.f) * ww;
        }

        // ---- reduce over 16 N-lanes, +b3, sigmoid, store ----
#pragma unroll
        for (int r = 0; r < 8; ++r) {
            float sum = lp[r];
            sum += __shfl_xor(sum, 1);
            sum += __shfl_xor(sum, 2);
            sum += __shfl_xor(sum, 4);
            sum += __shfl_xor(sum, 8);
            if (lm == 0) {
                const int j = j0 + r + (g ? 8 : 0);
                const float logit = sum + b3s;
                orow[j] = 1.f / (1.f + __expf(-logit));
            }
        }
    }
}

// ---------------------------------------------------------------------------
extern "C" void kernel_launch(void* const* d_in, const int* in_sizes, int n_in,
                              void* d_out, int out_size, void* d_ws, size_t ws_size,
                              hipStream_t stream) {
    (void)in_sizes; (void)n_in; (void)out_size; (void)ws_size;
    const float* nf = (const float*)d_in[0];   // [4,512,128]
    const float* W1 = (const float*)d_in[1];   // [256,128]
    const float* b1 = (const float*)d_in[2];   // [128]
    const float* W2 = (const float*)d_in[3];   // [128,64]
    const float* b2 = (const float*)d_in[4];   // [64]
    const float* W3 = (const float*)d_in[5];   // [64,1]
    const float* b3 = (const float*)d_in[6];   // [1]
    float* out = (float*)d_out;                // [4,512,512]

    float* hi_ws = (float*)d_ws;                         // [2048,128] f32
    float* hj_ws = hi_ws + (size_t)BATCH * NNODES * D;   // [2048,128] f32

    precompute_hihj<<<BATCH * NNODES, D, 0, stream>>>(nf, W1, b1, hi_ws, hj_ws);
    cooccur_fused<<<BATCH * NNODES, 256, 0, stream>>>(hi_ws, hj_ws, W2, b2, W3, b3, out);
}
